// ProbSparseAttention_17205638988528
// MI455X (gfx1250) — matrix operand order repaired and checked by reading, not compiled, not run on hardware
//
#include <hip/hip_runtime.h>
#include <math.h>

// ---------------- types & helpers ----------------
typedef __attribute__((ext_vector_type(16))) __bf16 v16bf;
typedef __attribute__((ext_vector_type(8)))  float  v8f;

// Native conversion: clang lowers f32->bf16 casts to hardware cvt ops
// (packed v_cvt_pk_bf16_f32 where possible), RNE rounding.
__device__ __forceinline__ __bf16 f2bf(float f) { return (__bf16)f; }

#define BB   2
#define LL   2048
#define DD   1024
#define HH   16
#define HD   64
#define UU   8
#define SCALE 0.03125f   // 1/sqrt(1024)

// =====================================================================
// Kernel 1: S[bh,q,k] = (Q K^T) * scale   via v_wmma_f32_16x16x32_bf16
// one wave -> 32(M) x 64(N) tile; block = 8 waves covering 32 x 512
// grid: (L/512, L/32, B*H)
// =====================================================================
__global__ void qk_scores_kernel(const float* __restrict__ Q,
                                 const float* __restrict__ K,
                                 float* __restrict__ S) {
    const int lane  = threadIdx.x & 31;
    const int wave  = threadIdx.x >> 5;
    const int bh    = blockIdx.z;
    const int b     = bh >> 4;
    const int h     = bh & 15;
    const int qbase = blockIdx.y * 32;
    const int nbase = blockIdx.x * 512 + wave * 64;

    const int mrow  = lane & 15;
    const int khalf = lane >> 4;

    const float* __restrict__ Qb = Q + (size_t)b * LL * DD + (size_t)h * HD;
    const float* __restrict__ Kb = K + (size_t)b * LL * DD + (size_t)h * HD;

    const float* qa0 = Qb + (size_t)(qbase + mrow)      * DD;  // A rows 0..15
    const float* qa1 = Qb + (size_t)(qbase + 16 + mrow) * DD;  // A rows 16..31
    const float* kb[4];
#pragma unroll
    for (int ct = 0; ct < 4; ++ct)
        kb[ct] = Kb + (size_t)(nbase + ct * 16 + mrow) * DD;   // B col tiles

    v8f c0[4] = {}, c1[4] = {};

#pragma unroll
    for (int k0 = 0; k0 < HD; k0 += 32) {
        v16bf a0, a1;
#pragma unroll
        for (int j = 0; j < 8; ++j) {
            a0[j]     = f2bf(qa0[k0 + khalf * 8 + j]      * SCALE);
            a0[8 + j] = f2bf(qa0[k0 + 16 + khalf * 8 + j] * SCALE);
            a1[j]     = f2bf(qa1[k0 + khalf * 8 + j]      * SCALE);
            a1[8 + j] = f2bf(qa1[k0 + 16 + khalf * 8 + j] * SCALE);
        }
#pragma unroll
        for (int ct = 0; ct < 4; ++ct) {
            v16bf bt;
#pragma unroll
            for (int j = 0; j < 16; ++j)
                bt[j] = f2bf(kb[ct][k0 + (j >> 3) * 16 + khalf * 8 + (j & 7)]);
            c0[ct] = __builtin_amdgcn_wmma_f32_16x16x32_bf16(false, a0, false, bt, (short)0, c0[ct], false, false);
            c1[ct] = __builtin_amdgcn_wmma_f32_16x16x32_bf16(false, a1, false, bt, (short)0, c1[ct], false, false);
        }
    }

    // store: VGPR i of C holds M = i + (lane>=16 ? 8 : 0), N = lane&15
    float* Sb = S + (size_t)bh * LL * LL;
#pragma unroll
    for (int ct = 0; ct < 4; ++ct) {
        const int col = nbase + ct * 16 + (lane & 15);
#pragma unroll
        for (int i = 0; i < 8; ++i) {
            const int r0 = qbase + i + khalf * 8;
            Sb[(size_t)r0 * LL + col]        = c0[ct][i];
            Sb[(size_t)(r0 + 16) * LL + col] = c1[ct][i];
        }
    }
}

// =====================================================================
// Kernel 2: per (b,h,q) row: top-8, softmax, rewrite row in place,
//           sparse attn @ V -> X[b,q,h*64+d]
// one wave32 per row; block = 8 waves; grid.x = B*H*L/8
// =====================================================================
__global__ void topk_softmax_av_kernel(const float* __restrict__ V,
                                       float* __restrict__ S,
                                       float* __restrict__ X) {
    const int lane = threadIdx.x & 31;
    const int gw   = blockIdx.x * 8 + (threadIdx.x >> 5);
    const int q    = gw & (LL - 1);
    const int bh   = gw >> 11;
    const int b    = bh >> 4;
    const int h    = bh & 15;

    float* __restrict__ row = S + ((size_t)bh * LL + q) * LL;

    // ---- lane-local top-8 over 64 elements (replace-min) ----
    float tv[UU]; int ti[UU];
#pragma unroll
    for (int s = 0; s < UU; ++s) { tv[s] = -1e30f; ti[s] = -1; }
    float curmin = -1e30f; int minpos = 0;

    const float4* row4 = (const float4*)row;
#pragma unroll 4
    for (int t = 0; t < 16; ++t) {
        float4 v4 = row4[t * 32 + lane];
        const int base = (t * 32 + lane) * 4;
        float vals[4] = {v4.x, v4.y, v4.z, v4.w};
#pragma unroll
        for (int j = 0; j < 4; ++j) {
            float val = vals[j];
            if (val > curmin) {
                tv[minpos] = val; ti[minpos] = base + j;
                curmin = tv[0]; minpos = 0;
#pragma unroll
                for (int s = 1; s < UU; ++s)
                    if (tv[s] < curmin) { curmin = tv[s]; minpos = s; }
            }
        }
    }

    // ---- 8 rounds of wave-wide argmax -> global top-8 (descending) ----
    float topv[UU]; int topi[UU];
#pragma unroll
    for (int r = 0; r < UU; ++r) {
        float lv = tv[0]; int li = ti[0];
#pragma unroll
        for (int s = 1; s < UU; ++s)
            if (tv[s] > lv) { lv = tv[s]; li = ti[s]; }
#pragma unroll
        for (int off = 16; off > 0; off >>= 1) {
            float ov = __shfl_xor(lv, off, 32);
            int   oi = __shfl_xor(li, off, 32);
            if (ov > lv || (ov == lv && oi < li)) { lv = ov; li = oi; }
        }
        topv[r] = lv; topi[r] = li;
#pragma unroll
        for (int s = 0; s < UU; ++s)
            if (ti[s] == li) tv[s] = -1e30f;   // only owner lane matches
    }

    // ---- softmax over the 8 kept scores ----
    const float m = topv[0];
    float w[UU]; float ssum = 0.f;
#pragma unroll
    for (int r = 0; r < UU; ++r) { w[r] = __expf(topv[r] - m); ssum += w[r]; }
    const float inv = 1.0f / ssum;
#pragma unroll
    for (int r = 0; r < UU; ++r) w[r] *= inv;

    // ---- rewrite row: zeros then scatter the 8 weights ----
    float4* row4w = (float4*)row;
    const float4 z = {0.f, 0.f, 0.f, 0.f};
#pragma unroll 4
    for (int t = 0; t < 16; ++t) row4w[t * 32 + lane] = z;
#pragma unroll
    for (int r = 0; r < UU; ++r)
        if (lane == r) row[topi[r]] = w[r];

    // ---- sparse attn @ V : out[d] = sum_r w[r] * V[b, topi[r], h*64+d] ----
    const float* __restrict__ Vb = V + (size_t)b * LL * DD + (size_t)h * HD;
    float acc0 = 0.f, acc1 = 0.f;
#pragma unroll
    for (int r = 0; r < UU; ++r) {
        const float* vr = Vb + (size_t)topi[r] * DD;
        acc0 += w[r] * vr[lane];
        acc1 += w[r] * vr[lane + 32];
    }
    float* xr = X + ((size_t)b * LL + q) * DD + (size_t)h * HD;
    xr[lane]      = acc0;
    xr[lane + 32] = acc1;
}

// =====================================================================
// Kernel 3: Y = X @ W^T + bias   (M=4096, N=1024, K=1024), bf16 WMMA
// one wave -> 32(M) x 64(N) tile; block = 8 waves (32 x 512); grid: (2, 128)
// =====================================================================
__global__ void out_proj_kernel(const float* __restrict__ X,
                                const float* __restrict__ W,
                                const float* __restrict__ bias,
                                float* __restrict__ Y) {
    const int lane  = threadIdx.x & 31;
    const int wave  = threadIdx.x >> 5;
    const int mbase = blockIdx.y * 32;
    const int nbase = blockIdx.x * 512 + wave * 64;
    const int mrow  = lane & 15;
    const int khalf = lane >> 4;

    const float* xa0 = X + (size_t)(mbase + mrow)      * DD;
    const float* xa1 = X + (size_t)(mbase + 16 + mrow) * DD;
    const float* wb[4];
#pragma unroll
    for (int ct = 0; ct < 4; ++ct)
        wb[ct] = W + (size_t)(nbase + ct * 16 + mrow) * DD;

    v8f c0[4] = {}, c1[4] = {};

    for (int kc = 0; kc < DD; kc += 32) {
        v16bf a0, a1;
#pragma unroll
        for (int j = 0; j < 8; ++j) {
            a0[j]     = f2bf(xa0[kc + khalf * 8 + j]);
            a0[8 + j] = f2bf(xa0[kc + 16 + khalf * 8 + j]);
            a1[j]     = f2bf(xa1[kc + khalf * 8 + j]);
            a1[8 + j] = f2bf(xa1[kc + 16 + khalf * 8 + j]);
        }
#pragma unroll
        for (int ct = 0; ct < 4; ++ct) {
            v16bf bt;
#pragma unroll
            for (int j = 0; j < 16; ++j)
                bt[j] = f2bf(wb[ct][kc + (j >> 3) * 16 + khalf * 8 + (j & 7)]);
            c0[ct] = __builtin_amdgcn_wmma_f32_16x16x32_bf16(false, a0, false, bt, (short)0, c0[ct], false, false);
            c1[ct] = __builtin_amdgcn_wmma_f32_16x16x32_bf16(false, a1, false, bt, (short)0, c1[ct], false, false);
        }
    }

#pragma unroll
    for (int ct = 0; ct < 4; ++ct) {
        const int col = nbase + ct * 16 + (lane & 15);
        const float bv = bias[col];
#pragma unroll
        for (int i = 0; i < 8; ++i) {
            const int r0 = mbase + i + khalf * 8;
            Y[(size_t)r0 * DD + col]        = c0[ct][i] + bv;
            Y[(size_t)(r0 + 16) * DD + col] = c1[ct][i] + bv;
        }
    }
}

// =====================================================================
extern "C" void kernel_launch(void* const* d_in, const int* in_sizes, int n_in,
                              void* d_out, int out_size, void* d_ws, size_t ws_size,
                              hipStream_t stream) {
    const float* Q    = (const float*)d_in[0];
    const float* K    = (const float*)d_in[1];
    const float* V    = (const float*)d_in[2];
    const float* W    = (const float*)d_in[3];
    const float* bias = (const float*)d_in[4];

    float* out      = (float*)d_out;
    float* attn_out = out;                                   // [B, L, D]
    float* attn_w   = out + (size_t)BB * LL * DD;            // [B, H, L, L]
    float* Xbuf     = (float*)d_ws;                          // [B, L, D] scratch (16 MB)

    // 1) scores = scale * Q K^T  (written straight into attn_weights region)
    dim3 g1(LL / 512, LL / 32, BB * HH);
    qk_scores_kernel<<<g1, 256, 0, stream>>>(Q, K, attn_w);

    // 2) top-8 + softmax (in-place row rewrite) + sparse attn@V -> Xbuf
    topk_softmax_av_kernel<<<(BB * HH * LL) / 8, 256, 0, stream>>>(V, attn_w, Xbuf);

    // 3) attn_out = Xbuf @ W^T + bias
    dim3 g3(DD / 512, (BB * LL) / 32, 1);
    out_proj_kernel<<<g3, 256, 0, stream>>>(Xbuf, W, bias, attn_out);
}
